// GNNGM1_94489280749
// MI455X (gfx1250) — matrix operand order repaired and checked by reading, not compiled, not run on hardware
//
#include <hip/hip_runtime.h>
#include <hip/hip_bf16.h>

#define NV  1024
#define HD  64

typedef __attribute__((ext_vector_type(16))) _Float16 v16h;
typedef __attribute__((ext_vector_type(8)))  float    v8f;

// ---------------------------------------------------------------- utilities
__global__ void zero_f32(float* __restrict__ p, int n) {
    int i = blockIdx.x * 256 + threadIdx.x;
    if (i < n) p[i] = 0.0f;
}

// --------------------------------------------------- x = MLP(rowsum(G)+1)
// one block (64 threads) per row; rows [0,1024) -> G1/x1, [1024,2048) -> G2/x2
__global__ void init_x_kernel(const float* __restrict__ G1, const float* __restrict__ G2,
                              const float* __restrict__ w1, const float* __restrict__ b1,
                              const float* __restrict__ w2, const float* __restrict__ b2,
                              float* __restrict__ x1, float* __restrict__ x2) {
    int row = blockIdx.x;
    const float* G = (row < NV) ? G1 : G2;
    float* xo      = (row < NV) ? x1 : x2;
    int r = (row < NV) ? row : row - NV;
    int t = threadIdx.x;                 // 64 threads
    __shared__ float sred[64];
    __shared__ float hsh[64];
    float s = 0.0f;
    for (int k = t; k < NV; k += 64) s += G[(size_t)r * NV + k];
    sred[t] = s;
    __syncthreads();
    for (int off = 32; off > 0; off >>= 1) {
        if (t < off) sred[t] += sred[t + off];
        __syncthreads();
    }
    float sv = sred[0] + 1.0f;           // +1 from the identity diagonal
    float h = sv * w1[t] + b1[t];
    hsh[t] = h > 0.0f ? h : 0.0f;
    __syncthreads();
    float acc = b2[t];
    #pragma unroll 8
    for (int k = 0; k < 64; ++k) acc += hsh[k] * w2[k * 64 + t];
    xo[(size_t)r * HD + t] = acc;
}

// ------------------------------------------------- per-pair readout MLP (WMMA)
// logit(i,j) = b2 + sum_n relu( P_ij*W1[0,n] + sum_k d_ijk*W1[1+k,n] + b1[n] ) * W2[n]
// d_ijk = -|x1[i,k]-x2[j,k]|/10.  Diff contraction done with v_wmma_f32_16x16x32_f16.
__global__ void __launch_bounds__(256)
readout_kernel(const float* __restrict__ x1, const float* __restrict__ x2,
               const float* __restrict__ P,
               const float* __restrict__ rw1,   // 65 x 64
               const float* __restrict__ rb1,   // 64
               const float* __restrict__ rw2,   // 64 (x1)
               const float* __restrict__ rb2,   // 1
               float* __restrict__ logits) {
    __shared__ _Float16 w1h[64][64];     // [k][n] = W1[1+k][n]
    __shared__ float w10[64], b1s[64], w2s[64];
    __shared__ float x1s[8][64];
    __shared__ float x2s[16][68];        // padded to dodge bank conflicts
    __shared__ float ptile[8][16];

    const int tid  = threadIdx.x;
    const int w    = tid >> 5;           // wave id: 0..7, one row i per wave
    const int lane = tid & 31;
    const int lo   = lane & 15;
    const int hi   = lane >> 4;
    const int i    = blockIdx.x * 8 + w;

    for (int idx = tid; idx < 64 * 64; idx += 256) {
        int k = idx >> 6, n = idx & 63;
        w1h[k][n] = (_Float16)rw1[(1 + k) * 64 + n];
    }
    if (tid < 64) {
        w10[tid] = rw1[tid];
        b1s[tid] = rb1[tid];
        w2s[tid] = rw2[tid];
    }
    for (int idx = tid; idx < 8 * 64; idx += 256) {
        int ww = idx >> 6, k = idx & 63;
        x1s[ww][k] = x1[(size_t)(blockIdx.x * 8 + ww) * HD + k];
    }
    __syncthreads();

    // B tiles resident in registers: B is KxN (32x16) f16; lanes 0-15 K=0..15,
    // lanes 16-31 K=16..31, N = lane%16  (2 halfs per VGPR).
    v16h Breg[2][4];
    #pragma unroll
    for (int kc = 0; kc < 2; ++kc) {
        #pragma unroll
        for (int t = 0; t < 4; ++t) {
            int n = t * 16 + lo;
            #pragma unroll
            for (int q = 0; q < 16; ++q) {
                int K = kc * 32 + hi * 16 + q;
                Breg[kc][t][q] = w1h[K][n];
            }
        }
    }
    float w2r[4], b1r[4], w10r[4];
    #pragma unroll
    for (int t = 0; t < 4; ++t) {
        int n = t * 16 + lo;
        w2r[t] = w2s[n]; b1r[t] = b1s[n]; w10r[t] = w10[n];
    }
    const float bias2 = rb2[0];

    const int jt0 = blockIdx.y * 16;
    for (int jt = jt0; jt < jt0 + 16; ++jt) {
        const int j0 = jt * 16;
        __syncthreads();
        for (int idx = tid; idx < 16 * 64; idx += 256) {
            int m = idx >> 6, k = idx & 63;
            x2s[m][k] = x2[(size_t)(j0 + m) * HD + k];
        }
        if (tid < 128) {
            int ww = tid >> 4, m = tid & 15;
            ptile[ww][m] = P[(size_t)(blockIdx.x * 8 + ww) * NV + j0 + m];
        }
        __syncthreads();

        v8f c[4];
        const v8f vzero = {0.f, 0.f, 0.f, 0.f, 0.f, 0.f, 0.f, 0.f};
        #pragma unroll
        for (int t = 0; t < 4; ++t) c[t] = vzero;

        #pragma unroll
        for (int kc = 0; kc < 2; ++kc) {
            // A is MxK (16x32) f16: M = lane%16 (pair m), per-half K pattern:
            // half0: K in {0..15}, half1: {16..31}
            v16h a;
            #pragma unroll
            for (int q = 0; q < 16; ++q) {
                int K  = (hi << 4) + q;
                int kd = kc * 32 + K;
                float d = -fabsf(x1s[w][kd] - x2s[lo][kd]) * 0.1f;
                a[q] = (_Float16)d;
            }
            #pragma unroll
            for (int t = 0; t < 4; ++t)
                c[t] = __builtin_amdgcn_wmma_f32_16x16x32_f16(
                        false, a, false, Breg[kc][t], (short)0, c[t], false, false);
        }

        // epilogue: add rank-1 P term + bias, relu, dot with W2, reduce across N
        float part[8];
        #pragma unroll
        for (int r = 0; r < 8; ++r) part[r] = 0.0f;
        #pragma unroll
        for (int t = 0; t < 4; ++t) {
            #pragma unroll
            for (int r = 0; r < 8; ++r) {
                float pre = c[t][r] + ptile[w][hi * 8 + r] * w10r[t] + b1r[t];
                float hv  = pre > 0.0f ? pre : 0.0f;
                part[r] += hv * w2r[t];
            }
        }
        #pragma unroll
        for (int r = 0; r < 8; ++r) {
            #pragma unroll
            for (int mask = 1; mask <= 8; mask <<= 1)
                part[r] += __shfl_xor(part[r], mask, 32);
        }
        if (lo == 0) {
            #pragma unroll
            for (int r = 0; r < 8; ++r)
                logits[(size_t)i * NV + j0 + hi * 8 + r] = part[r] + bias2;
        }
    }
}

// ---------------------------------------- in-place column softmax + nan_to_num
__global__ void softmax_col_kernel(float* __restrict__ S) {
    __shared__ float red[16][17];
    __shared__ float cmax[16];
    __shared__ float csum[16];
    const float NEG_INF = -__builtin_inff();
    int tx = threadIdx.x, ty = threadIdx.y;
    int j = blockIdx.x * 16 + tx;
    float m = NEG_INF;
    for (int r = ty; r < NV; r += 16) {
        float v = S[(size_t)r * NV + j];
        if (v != v) v = NEG_INF;         // NaN -> -inf, so that entry becomes 0
        m = fmaxf(m, v);
    }
    red[ty][tx] = m;
    __syncthreads();
    for (int off = 8; off > 0; off >>= 1) {
        if (ty < off) red[ty][tx] = fmaxf(red[ty][tx], red[ty + off][tx]);
        __syncthreads();
    }
    if (ty == 0) cmax[tx] = red[0][tx];
    __syncthreads();
    float cm = cmax[tx];
    float s = 0.0f;
    for (int r = ty; r < NV; r += 16) {
        float v = S[(size_t)r * NV + j];
        if (v != v) v = NEG_INF;
        if (cm != NEG_INF) s += __expf(v - cm);
    }
    red[ty][tx] = s;
    __syncthreads();
    for (int off = 8; off > 0; off >>= 1) {
        if (ty < off) red[ty][tx] += red[ty + off][tx];
        __syncthreads();
    }
    if (ty == 0) csum[tx] = red[0][tx];
    __syncthreads();
    float inv = (cm == NEG_INF || csum[tx] <= 0.0f) ? 0.0f : 1.0f / csum[tx];
    for (int r = ty; r < NV; r += 16) {
        float v = S[(size_t)r * NV + j];
        if (v != v) v = NEG_INF;
        float e = (cm == NEG_INF) ? 0.0f : __expf(v - cm);
        S[(size_t)r * NV + j] = e * inv;
    }
}

// ------------------------------------------------------- W = log(1536 * S)
__global__ void wlog_kernel(const float* __restrict__ S, float* __restrict__ W, int n) {
    int idx = blockIdx.x * 256 + threadIdx.x;
    if (idx < n) W[idx] = __logf(1536.0f * S[idx]);
}

// -------------------------------------------- 4x4 LAP by 24-permutation max
__global__ void lap_kernel(const float* __restrict__ W, const int* __restrict__ adj1,
                           const int* __restrict__ adj2, float* __restrict__ Pout) {
    int idx = blockIdx.x * 256 + threadIdx.x;
    if (idx >= NV * NV) return;
    int i = idx >> 10, j = idx & (NV - 1);
    int a1[4], a2[4];
    #pragma unroll
    for (int k = 0; k < 4; ++k) { a1[k] = adj1[i * 4 + k]; a2[k] = adj2[j * 4 + k]; }
    float sub[4][4];
    #pragma unroll
    for (int a = 0; a < 4; ++a)
        #pragma unroll
        for (int b = 0; b < 4; ++b)
            sub[a][b] = W[(size_t)a1[a] * NV + a2[b]];
    static constexpr int PERMS[24][4] = {
        {0,1,2,3},{0,1,3,2},{0,2,1,3},{0,2,3,1},{0,3,1,2},{0,3,2,1},
        {1,0,2,3},{1,0,3,2},{1,2,0,3},{1,2,3,0},{1,3,0,2},{1,3,2,0},
        {2,0,1,3},{2,0,3,1},{2,1,0,3},{2,1,3,0},{2,3,0,1},{2,3,1,0},
        {3,0,1,2},{3,0,2,1},{3,1,0,2},{3,1,2,0},{3,2,0,1},{3,2,1,0}};
    float best = -__builtin_inff();
    #pragma unroll
    for (int p = 0; p < 24; ++p) {
        float s = sub[0][PERMS[p][0]] + sub[1][PERMS[p][1]]
                + sub[2][PERMS[p][2]] + sub[3][PERMS[p][3]];
        best = fmaxf(best, s);
    }
    Pout[idx] = best;
}

// ------------------------------------------------------------- r2 = S^T @ r1
__global__ void r2_kernel(const float* __restrict__ S, const float* __restrict__ r1,
                          float* __restrict__ r2) {
    __shared__ float buf[64][6];
    int j = blockIdx.x, t = threadIdx.x;     // 64 threads
    float acc[6] = {0, 0, 0, 0, 0, 0};
    for (int i = t; i < NV; i += 64) {
        float s = S[(size_t)i * NV + j];
        #pragma unroll
        for (int d = 0; d < 6; ++d) acc[d] += s * r1[i * 6 + d];
    }
    #pragma unroll
    for (int d = 0; d < 6; ++d) buf[t][d] = acc[d];
    __syncthreads();
    for (int off = 32; off > 0; off >>= 1) {
        if (t < off) {
            #pragma unroll
            for (int d = 0; d < 6; ++d) buf[t][d] += buf[t + off][d];
        }
        __syncthreads();
    }
    if (t < 6) r2[j * 6 + t] = buf[0][t];
}

// ----------------------------------------- y = (G+I) @ concat(x[1024x64], r[1024x6])
__global__ void y_kernel(const float* __restrict__ G, const float* __restrict__ x,
                         const float* __restrict__ r, float* __restrict__ y) {
    __shared__ float gch[128];
    int i = blockIdx.x, t = threadIdx.x;     // 128 threads, outputs for t<70
    int c = t;
    float acc = 0.0f;
    if (c < 70) acc = (c < 64) ? x[(size_t)i * 64 + c] : r[i * 6 + (c - 64)]; // identity
    for (int base = 0; base < NV; base += 128) {
        __syncthreads();
        gch[t] = G[(size_t)i * NV + base + t];
        __syncthreads();
        if (c < 70) {
            for (int kk = 0; kk < 128; ++kk) {
                float g = gch[kk];
                if (g != 0.0f) {          // 1%-dense adjacency: skip zeros
                    int k = base + kk;
                    float xv = (c < 64) ? x[(size_t)k * 64 + c] : r[k * 6 + (c - 64)];
                    acc += g * xv;
                }
            }
        }
    }
    if (c < 70) y[(size_t)i * 70 + c] = acc;
}

// ------------------------------------------------- x = MLP70(y): 70 -> 64 -> 64
__global__ void mlp70_kernel(const float* __restrict__ y, const float* __restrict__ w1,
                             const float* __restrict__ b1, const float* __restrict__ w2,
                             const float* __restrict__ b2, float* __restrict__ xo) {
    __shared__ float yrow[70];
    __shared__ float hsh[64];
    int i = blockIdx.x, t = threadIdx.x;     // 64 threads
    for (int c = t; c < 70; c += 64) yrow[c] = y[(size_t)i * 70 + c];
    __syncthreads();
    float h = b1[t];
    #pragma unroll 7
    for (int c = 0; c < 70; ++c) h += yrow[c] * w1[c * 64 + t];
    hsh[t] = h > 0.0f ? h : 0.0f;
    __syncthreads();
    float acc = b2[t];
    #pragma unroll 8
    for (int k = 0; k < 64; ++k) acc += hsh[k] * w2[k * 64 + t];
    xo[(size_t)i * 64 + t] = acc;
}

// ------------------------------------------------------------------ driver
extern "C" void kernel_launch(void* const* d_in, const int* in_sizes, int n_in,
                              void* d_out, int out_size, void* d_ws, size_t ws_size,
                              hipStream_t stream) {
    (void)in_sizes; (void)n_in; (void)out_size; (void)ws_size;
    const float* G1    = (const float*)d_in[0];
    const float* G2    = (const float*)d_in[1];
    const float* r_all = (const float*)d_in[2];   // (1,1024,6)
    const float* iw1   = (const float*)d_in[3];
    const float* ib1   = (const float*)d_in[4];
    const float* iw2   = (const float*)d_in[5];
    const float* ib2   = (const float*)d_in[6];
    const float* rw1   = (const float*)d_in[7];   // (2,65,64)
    const float* rb1   = (const float*)d_in[8];   // (2,64)
    const float* rw2   = (const float*)d_in[9];   // (2,64,1)
    const float* rb2   = (const float*)d_in[10];  // (2,1)
    const float* mw1   = (const float*)d_in[11];  // (1,70,64)
    const float* mb1   = (const float*)d_in[12];
    const float* mw2   = (const float*)d_in[13];
    const float* mb2   = (const float*)d_in[14];
    const int*   adj1  = (const int*)d_in[15];
    const int*   adj2  = (const int*)d_in[16];
    float* out = (float*)d_out;                   // [P, S0, S1] : 3 x 1024 x 1024

    float* ws  = (float*)d_ws;
    float* Pws = ws;                     // 1M
    float* Wws = Pws + NV * NV;          // 1M
    float* x1  = Wws + NV * NV;          // 64K
    float* x2  = x1 + NV * HD;
    float* y1  = x2 + NV * HD;           // 1024*70
    float* y2  = y1 + NV * 70;
    float* r2  = y2 + NV * 70;           // 1024*6

    const int npix = NV * NV;
    zero_f32<<<(npix + 255) / 256, 256, 0, stream>>>(Pws, npix);
    init_x_kernel<<<2 * NV, 64, 0, stream>>>(G1, G2, iw1, ib1, iw2, ib2, x1, x2);

    // -------- layer 0
    float* S0 = out + npix;
    readout_kernel<<<dim3(NV / 8, 4), 256, 0, stream>>>(x1, x2, Pws, rw1, rb1, rw2, rb2, S0);
    softmax_col_kernel<<<NV / 16, dim3(16, 16), 0, stream>>>(S0);
    wlog_kernel<<<(npix + 255) / 256, 256, 0, stream>>>(S0, Wws, npix);
    lap_kernel<<<(npix + 255) / 256, 256, 0, stream>>>(Wws, adj1, adj2, Pws);
    r2_kernel<<<NV, 64, 0, stream>>>(S0, r_all, r2);
    y_kernel<<<NV, 128, 0, stream>>>(G1, x1, r_all, y1);
    y_kernel<<<NV, 128, 0, stream>>>(G2, x2, r2, y2);
    mlp70_kernel<<<NV, 64, 0, stream>>>(y1, mw1, mb1, mw2, mb2, x1);
    mlp70_kernel<<<NV, 64, 0, stream>>>(y2, mw1, mb1, mw2, mb2, x2);

    // -------- layer 1
    float* S1 = out + 2 * npix;
    readout_kernel<<<dim3(NV / 8, 4), 256, 0, stream>>>(x1, x2, Pws,
            rw1 + 65 * 64, rb1 + 64, rw2 + 64, rb2 + 1, S1);
    softmax_col_kernel<<<NV / 16, dim3(16, 16), 0, stream>>>(S1);
    wlog_kernel<<<(npix + 255) / 256, 256, 0, stream>>>(S1, Wws, npix);
    lap_kernel<<<(npix + 255) / 256, 256, 0, stream>>>(Wws, adj1, adj2, out); // final P
}
